// SegBiGRU_72284299592178
// MI455X (gfx1250) — matrix operand order repaired and checked by reading, not compiled, not run on hardware
//
#include <hip/hip_runtime.h>

typedef _Float16 f16;
typedef __attribute__((ext_vector_type(16))) _Float16 v16h;
typedef __attribute__((ext_vector_type(8)))  float    v8f;

#define B_  2048
#define T_  40
#define H_  128
#define C_  102
#define G3_ 384   // 3*H
#define CP_ 112   // classes padded to 7*16

// Padded LDS row strides (avoid 256B-period bank conflicts; 272B = 4-bank rotate/row)
#define LDW_ 136  // halves, W_hh tile rows
#define LDH_ 136  // floats, h-state rows
#define LDA_ 136  // halves, mix/hidden tiles
#define LDP_ 72   // halves, P2 rows (144B = 36-bank rotate)

// ---------------------------------------------------------------- WMMA helpers
__device__ __forceinline__ v8f wmma_f16(v16h a, v16h b, v8f c) {
  // D = A(16x32) x B(32x16) + C, f32 accumulate
  return __builtin_amdgcn_wmma_f32_16x16x32_f16(false, a, false, b, (short)0, c,
                                                false, false);
}

// A fragment (16xK row-major f16 source). ISA 16-bit A 16x32 layout:
// lane m=lane&15; comps0-7 -> K=kbase+(lane>=16?8:0)+i; comps8-15 -> +16.
__device__ __forceinline__ v16h load_a_f16(const f16* src, int ld, int kbase, int lane) {
  const f16* row = src + (lane & 15) * ld + kbase + ((lane >> 4) << 3);
  v16h a;
#pragma unroll
  for (int i = 0; i < 8; ++i) a[i] = row[i];
#pragma unroll
  for (int i = 0; i < 8; ++i) a[8 + i] = row[16 + i];
  return a;
}

// Same layout, f32 source with conversion (for LDS-resident h state).
__device__ __forceinline__ v16h load_a_f32(const float* src, int ld, int kbase, int lane) {
  const float* row = src + (lane & 15) * ld + kbase + ((lane >> 4) << 3);
  v16h a;
#pragma unroll
  for (int i = 0; i < 8; ++i) a[i] = (f16)row[i];
#pragma unroll
  for (int i = 0; i < 8; ++i) a[8 + i] = (f16)row[16 + i];
  return a;
}

// Same layout, per-lane gathered row pointer (embedding gather).
__device__ __forceinline__ v16h load_a_f32_row(const float* row, int kbase, int lane) {
  const float* p = row + kbase + ((lane >> 4) << 3);
  v16h a;
#pragma unroll
  for (int i = 0; i < 8; ++i) a[i] = (f16)p[i];
#pragma unroll
  for (int i = 0; i < 8; ++i) a[8 + i] = (f16)p[16 + i];
  return a;
}

// B fragment where B[k][n] = S[n][k] (S row-major). ISA 16-bit B 32x16:
// n = lane&15; comp c -> K = kbase + (lane>>4)*8 + (c&7) + (c>>3)*16.
__device__ __forceinline__ v16h load_b_tr(const f16* S, int ld, int kbase, int lane) {
  const f16* row = S + (lane & 15) * ld + kbase + ((lane >> 4) << 3);
  v16h b;
#pragma unroll
  for (int i = 0; i < 16; ++i) b[i] = row[(i & 7) + ((i >> 3) << 4)];
  return b;
}

// B fragment where B[k][n] = S[k][n] (direct row-major).
__device__ __forceinline__ v16h load_b_dir(const f16* S, int ld, int kbase, int lane) {
  const f16* p = S + (kbase + ((lane >> 4) << 3)) * ld + (lane & 15);
  v16h b;
#pragma unroll
  for (int i = 0; i < 16; ++i) b[i] = p[((i & 7) + ((i >> 3) << 4)) * ld];
  return b;
}

// ---------------------------------------------------------------- prep: f32->f16 weights
__global__ void __launch_bounds__(256) k_prep(
    const float* __restrict__ wih, const float* __restrict__ whh,
    const float* __restrict__ wcls,
    f16* __restrict__ wih16, f16* __restrict__ whh16, f16* __restrict__ wcls16) {
  const int i = blockIdx.x * 256 + threadIdx.x;
  if (i < G3_ * H_) {
    wih16[i] = (f16)wih[i];
    whh16[i] = (f16)whh[i];
  }
  if (i < CP_ * H_) {                       // only first H cols of W_cls matter
    const int c = i >> 7, k = i & 127;
    wcls16[i] = (c < C_) ? (f16)wcls[c * 2 * H_ + k] : (f16)0.f;
  }
}

// ---------------------------------------------------------------- gx = gather(emb) @ W_ih^T + b_ih
// rows n = t*B + b ; output gx[n][384] f16. grid.x = 81920/16, grid.y = table(a/c)
__global__ void __launch_bounds__(256) k_gates(
    const int* __restrict__ tokens,
    const float* __restrict__ emb_a, const float* __restrict__ emb_c,
    const f16* __restrict__ wih16, const float* __restrict__ b_ih,
    f16* __restrict__ gxa, f16* __restrict__ gxc) {
  const int lane = threadIdx.x & 31;
  const int wave = threadIdx.x >> 5;
  const int n0 = blockIdx.x << 4;
  const int t  = n0 >> 11;                  // B_ == 2048
  const int b0 = n0 & (B_ - 1);
  const float* emb = blockIdx.y ? emb_c : emb_a;
  f16* gx = blockIdx.y ? gxc : gxa;

  const int tok = tokens[(b0 + (lane & 15)) * T_ + t];
  const float* xrow = emb + (size_t)tok * H_;
  v16h A[4];
#pragma unroll
  for (int ks = 0; ks < 4; ++ks) A[ks] = load_a_f32_row(xrow, ks * 32, lane);

  for (int j = 0; j < 3; ++j) {             // 8 waves * 3 = 24 n-tiles of 16 = 384
    const int g0 = (wave * 3 + j) << 4;
    v8f acc = {0.f, 0.f, 0.f, 0.f, 0.f, 0.f, 0.f, 0.f};
#pragma unroll
    for (int ks = 0; ks < 4; ++ks) {
      v16h bf = load_b_tr(wih16 + (size_t)g0 * H_, H_, ks * 32, lane);
      acc = wmma_f16(A[ks], bf, acc);
    }
    const int col = g0 + (lane & 15);
    const float bias = b_ih[col];
#pragma unroll
    for (int r = 0; r < 8; ++r) {
      const int m = r + ((lane >> 4) << 3);
      gx[(size_t)(n0 + m) * G3_ + col] = (f16)(acc[r] + bias);
    }
  }
}

// ---------------------------------------------------------------- GRU recurrence
// grid = (B/32 batch tiles, 4 scans). 128 threads. W_hh + h live in LDS for all 40 steps.
__global__ void __launch_bounds__(128) k_gru(
    const f16* __restrict__ gxa, const f16* __restrict__ gxc,
    const f16* __restrict__ whh16, const float* __restrict__ b_hh,
    f16* __restrict__ mixa, f16* __restrict__ mixc) {
  extern __shared__ char smem[];
  f16*   sW = (f16*)smem;                                   // [384][136]  104448 B
  float* sH = (float*)(smem + G3_ * LDW_ * sizeof(f16));    // [32][136]    17408 B

  const int scan = blockIdx.y;                              // 0:f_a 1:b_a 2:f_c 3:b_c
  const int rev  = scan & 1;
  const f16* gx  = (scan >> 1) ? gxc : gxa;
  f16* mix       = (scan >> 1) ? mixc : mixa;
  const int b0   = blockIdx.x << 5;

  {
    const uint4* src = (const uint4*)whh16;                 // 16 uint4 per 128-half row
    uint4* dst = (uint4*)sW;                                // 17 uint4 per padded row
    for (int i = threadIdx.x; i < G3_ * 16; i += 128) {
      const int row = i >> 4, q = i & 15;
      dst[row * 17 + q] = src[i];
    }
    for (int i = threadIdx.x; i < 32 * LDH_; i += 128) sH[i] = 0.f;
  }
  __syncthreads();

  const int lane = threadIdx.x & 31;
  const int wave = threadIdx.x >> 5;
  const int mr0  = (wave & 1) << 4;   // 16-row sub-tile
  const int j0   = (wave >> 1) << 6;  // 64-column half: wave owns all 3 gates for its j's

  float bias[3][4];
#pragma unroll
  for (int g = 0; g < 3; ++g)
#pragma unroll
    for (int jt = 0; jt < 4; ++jt)
      bias[g][jt] = b_hh[g * H_ + j0 + (jt << 4) + (lane & 15)];

  const v8f zero8 = {0.f, 0.f, 0.f, 0.f, 0.f, 0.f, 0.f, 0.f};

  for (int p = 0; p < T_; ++p) {
    const int t = rev ? (T_ - 1 - p) : p;

    v16h A[4];
#pragma unroll
    for (int ks = 0; ks < 4; ++ks)
      A[ks] = load_a_f32(sH + mr0 * LDH_, LDH_, ks * 32, lane);

    // Prefetch next step's gate rows (this wave's 48 cache lines) so the HBM/L2
    // latency overlaps the WMMA work below. Lowers to global_prefetch_b8.
    {
      const int pn = (p + 1 < T_) ? p + 1 : p;
      const int tn = rev ? (T_ - 1 - pn) : pn;
      const int m  = mr0 + ((lane >> 4) << 3);              // 2 row groups/lane half
#pragma unroll
      for (int r = 0; r < 8; ++r) {
        const f16* gpre = gx + ((size_t)tn * B_ + b0 + m + r) * G3_ + j0 + (lane & 15);
        __builtin_prefetch(gpre, 0, 1);
        __builtin_prefetch(gpre + H_, 0, 1);
        __builtin_prefetch(gpre + 2 * H_, 0, 1);
      }
    }
    __syncthreads();  // all waves finished reading h before anyone writes it

    v8f acc[3][4];
#pragma unroll
    for (int g = 0; g < 3; ++g)
#pragma unroll
      for (int jt = 0; jt < 4; ++jt) acc[g][jt] = zero8;

#pragma unroll
    for (int ks = 0; ks < 4; ++ks)
#pragma unroll
      for (int g = 0; g < 3; ++g)
#pragma unroll
        for (int jt = 0; jt < 4; ++jt) {
          v16h bf = load_b_tr(sW + (size_t)(g * H_ + j0 + (jt << 4)) * LDW_,
                              LDW_, ks * 32, lane);
          acc[g][jt] = wmma_f16(A[ks], bf, acc[g][jt]);
        }

    // buffer-overwrite quirk: fwd keeps t>=T/2, bwd keeps t<T/2
    const bool keep = rev ? (t < T_ / 2) : (t >= T_ / 2);
#pragma unroll
    for (int jt = 0; jt < 4; ++jt) {
      const int col = j0 + (jt << 4) + (lane & 15);
#pragma unroll
      for (int r = 0; r < 8; ++r) {
        const int m = mr0 + r + ((lane >> 4) << 3);
        const int bb = b0 + m;
        const f16* gr = gx + ((size_t)t * B_ + bb) * G3_;
        const float xr = (float)gr[col];
        const float xz = (float)gr[H_ + col];
        const float xn = (float)gr[2 * H_ + col];
        const float hr = acc[0][jt][r] + bias[0][jt];
        const float hz = acc[1][jt][r] + bias[1][jt];
        const float hn = acc[2][jt][r] + bias[2][jt];
        const float rg = 1.f / (1.f + __expf(-(xr + hr)));
        const float zg = 1.f / (1.f + __expf(-(xz + hz)));
        const float ng = tanhf(xn + rg * hn);
        const float hnew = (1.f - zg) * ng + zg * sH[m * LDH_ + col];
        sH[m * LDH_ + col] = hnew;
        if (keep) mix[((size_t)bb * T_ + t) * H_ + col] = (f16)hnew;
      }
    }
    __syncthreads();  // h writes visible before next step's reads
  }
}

// ---------------------------------------------------------------- per-batch attention + classifier
// M = mixa.mixa^T (48x48,K=128); col-softmax; hidden = P@mixc (K=64 pad);
// logits = hidden@Wcls^T (48x112,K=128); row-softmax -> out. One block per batch.
__global__ void __launch_bounds__(128) k_attn_cls(
    const f16* __restrict__ mixa, const f16* __restrict__ mixc,
    const f16* __restrict__ wcls16, const float* __restrict__ b_cls,
    float* __restrict__ out) {
  extern __shared__ char smem[];
  f16*   sA   = (f16*)(smem);             // [48][136] 13056 B
  f16*   sC   = (f16*)(smem + 13056);     // [48][136] 13056 B
  float* sM   = (float*)(smem + 26112);   // [48][48]   9216 B
  f16*   sP   = (f16*)(smem + 35328);     // [48][72]   6912 B (K padded to 64 + pad)
  f16*   sHid = (f16*)(smem + 42240);     // [48][136] 13056 B
  float* sLog = (float*)(smem + 55296);   // [48][112] 21504 B

  const int tid = threadIdx.x;
  const int lane = tid & 31, wave = tid >> 5;
  const int b = blockIdx.x;

  {
    const uint4* ga = (const uint4*)(mixa + (size_t)b * T_ * H_);
    const uint4* gc = (const uint4*)(mixc + (size_t)b * T_ * H_);
    uint4* da = (uint4*)sA; uint4* dc = (uint4*)sC; uint4* dp = (uint4*)sP;
    uint4 z; z.x = z.y = z.z = z.w = 0u;
    for (int i = tid; i < 48 * 16; i += 128) {      // 17 uint4 per padded row
      const int row = i >> 4, q = i & 15;
      uint4 va = z, vc = z;
      if (row < T_) { va = ga[row * 16 + q]; vc = gc[row * 16 + q]; }
      da[row * 17 + q] = va;
      dc[row * 17 + q] = vc;
    }
    for (int i = tid; i < (48 * LDP_) / 8; i += 128) dp[i] = z;  // zero P2 + padding
  }
  __syncthreads();

  // M tiles (3x3)
  for (int td = wave; td < 9; td += 4) {
    const int it = td / 3, jt = td % 3;
    v8f acc = {0.f, 0.f, 0.f, 0.f, 0.f, 0.f, 0.f, 0.f};
#pragma unroll
    for (int ks = 0; ks < 4; ++ks) {
      v16h a  = load_a_f16(sA + it * 16 * LDA_, LDA_, ks * 32, lane);
      v16h bf = load_b_tr(sA + jt * 16 * LDA_, LDA_, ks * 32, lane);
      acc = wmma_f16(a, bf, acc);
    }
    const int col = jt * 16 + (lane & 15);
#pragma unroll
    for (int r = 0; r < 8; ++r)
      sM[(it * 16 + r + ((lane >> 4) << 3)) * 48 + col] = acc[r];
  }
  __syncthreads();

  // softmax over rows i per column t -> P2[s][t] = exp(M[s][t]) / sum_i exp(M[i][t])
  if (tid < T_) {
    float mx = -1e30f;
    for (int i = 0; i < T_; ++i) mx = fmaxf(mx, sM[i * 48 + tid]);
    float s = 0.f;
    for (int i = 0; i < T_; ++i) {
      float e = __expf(sM[i * 48 + tid] - mx);
      sM[i * 48 + tid] = e;
      s += e;
    }
    const float inv = 1.f / s;
    for (int i = 0; i < T_; ++i) sP[i * LDP_ + tid] = (f16)(sM[i * 48 + tid] * inv);
  }
  __syncthreads();

  // hidden = P2 @ mixc (3x8 tiles, K=64)
  for (int td = wave; td < 24; td += 4) {
    const int st = td >> 3, nt = td & 7;
    v8f acc = {0.f, 0.f, 0.f, 0.f, 0.f, 0.f, 0.f, 0.f};
#pragma unroll
    for (int ks = 0; ks < 2; ++ks) {
      v16h a  = load_a_f16(sP + st * 16 * LDP_, LDP_, ks * 32, lane);
      v16h bf = load_b_dir(sC + (nt << 4), LDA_, ks * 32, lane);
      acc = wmma_f16(a, bf, acc);
    }
    const int col = (nt << 4) + (lane & 15);
#pragma unroll
    for (int r = 0; r < 8; ++r)
      sHid[(st * 16 + r + ((lane >> 4) << 3)) * LDA_ + col] = (f16)acc[r];
  }
  __syncthreads();

  // logits = hidden @ Wcls^T + b_cls (3x7 tiles, K=128)
  for (int td = wave; td < 21; td += 4) {
    const int st = td / 7, ct = td % 7;
    v8f acc = {0.f, 0.f, 0.f, 0.f, 0.f, 0.f, 0.f, 0.f};
#pragma unroll
    for (int ks = 0; ks < 4; ++ks) {
      v16h a  = load_a_f16(sHid + st * 16 * LDA_, LDA_, ks * 32, lane);
      v16h bf = load_b_tr(wcls16 + (size_t)(ct << 4) * H_, H_, ks * 32, lane);
      acc = wmma_f16(a, bf, acc);
    }
    const int c = (ct << 4) + (lane & 15);
    const float bias = (c < C_) ? b_cls[c] : 0.f;
#pragma unroll
    for (int r = 0; r < 8; ++r)
      sLog[(st * 16 + r + ((lane >> 4) << 3)) * CP_ + c] = acc[r] + bias;
  }
  __syncthreads();

  // row softmax over classes, write output row b*T + s
  if (tid < T_) {
    float mx = -1e30f;
    for (int c = 0; c < C_; ++c) mx = fmaxf(mx, sLog[tid * CP_ + c]);
    float s = 0.f;
    for (int c = 0; c < C_; ++c) s += __expf(sLog[tid * CP_ + c] - mx);
    const float inv = 1.f / s;
    float* o = out + ((size_t)b * T_ + tid) * C_;
    for (int c = 0; c < C_; ++c) o[c] = __expf(sLog[tid * CP_ + c] - mx) * inv;
  }
}

// ---------------------------------------------------------------- host
extern "C" void kernel_launch(void* const* d_in, const int* in_sizes, int n_in,
                              void* d_out, int out_size, void* d_ws, size_t ws_size,
                              hipStream_t stream) {
  (void)in_sizes; (void)n_in; (void)out_size; (void)ws_size;
  const int*   tokens = (const int*)d_in[0];
  const float* emb_c  = (const float*)d_in[1];
  const float* emb_a  = (const float*)d_in[2];
  const float* W_ih   = (const float*)d_in[3];
  const float* W_hh   = (const float*)d_in[4];
  const float* b_ih   = (const float*)d_in[5];
  const float* b_hh   = (const float*)d_in[6];
  const float* W_cls  = (const float*)d_in[7];
  const float* b_cls  = (const float*)d_in[8];
  float* out = (float*)d_out;

  char* ws = (char*)d_ws;
  size_t off = 0;
  auto take = [&](size_t bytes) {
    char* p = ws + off;
    off = (off + bytes + 255) & ~(size_t)255;
    return p;
  };
  f16* wih16  = (f16*)take((size_t)G3_ * H_ * sizeof(f16));
  f16* whh16  = (f16*)take((size_t)G3_ * H_ * sizeof(f16));
  f16* wcls16 = (f16*)take((size_t)CP_ * H_ * sizeof(f16));
  f16* gxa    = (f16*)take((size_t)B_ * T_ * G3_ * sizeof(f16));
  f16* gxc    = (f16*)take((size_t)B_ * T_ * G3_ * sizeof(f16));
  f16* mixa   = (f16*)take((size_t)B_ * T_ * H_ * sizeof(f16));
  f16* mixc   = (f16*)take((size_t)B_ * T_ * H_ * sizeof(f16));

  const int gru_lds  = G3_ * LDW_ * (int)sizeof(f16) +
                       32 * LDH_ * (int)sizeof(float);      // 121856
  const int attn_lds = 76800;
  (void)hipFuncSetAttribute((const void*)k_gru,
                            hipFuncAttributeMaxDynamicSharedMemorySize, gru_lds);
  (void)hipFuncSetAttribute((const void*)k_attn_cls,
                            hipFuncAttributeMaxDynamicSharedMemorySize, attn_lds);

  k_prep<<<(G3_ * H_ + 255) / 256, 256, 0, stream>>>(W_ih, W_hh, W_cls,
                                                     wih16, whh16, wcls16);
  k_gates<<<dim3((B_ * T_) / 16, 2), 256, 0, stream>>>(tokens, emb_a, emb_c,
                                                       wih16, b_ih, gxa, gxc);
  k_gru<<<dim3(B_ / 32, 4), 128, gru_lds, stream>>>(gxa, gxc, whh16, b_hh,
                                                    mixa, mixc);
  k_attn_cls<<<B_, 128, attn_lds, stream>>>(mixa, mixc, wcls16, b_cls, out);
}
